// MultiHeadSelfAttention_27075473834051
// MI455X (gfx1250) — matrix operand order repaired and checked by reading, not compiled, not run on hardware
//
#include <hip/hip_runtime.h>

#define TOKENS 4096
#define IN_DIM 1024
#define EMB 1024
#define NHEAD 16
#define HDIM 64
#define QKV_DIM 3072

typedef __attribute__((ext_vector_type(16))) __bf16 v16bf;
typedef __attribute__((ext_vector_type(8)))  float  v8f;

union Frag {
    v16bf v;
    uint4 q[2];
    unsigned short u[16];
};

__device__ __forceinline__ unsigned short f2bf(float f) {
    unsigned int u = __float_as_uint(f);
    return (unsigned short)((u + 0x7FFFu + ((u >> 16) & 1u)) >> 16);
}

__device__ __forceinline__ v8f wmma_bf16(const Frag& a, const Frag& b, v8f c) {
    return __builtin_amdgcn_wmma_f32_16x16x32_bf16(false, a.v, false, b.v,
                                                   (short)0, c, false, false);
}

// ---------------- fp32 -> bf16 conversion ----------------
__global__ void cvt_kernel(const float* __restrict__ src,
                           unsigned short* __restrict__ dst, int n) {
    int i = blockIdx.x * blockDim.x + threadIdx.x;
    if (i < n) dst[i] = f2bf(src[i]);
}

// ---------------- QKV projection: qkv = x @ qkv_w^T ----------------
// One wave computes a 16(M tokens) x 64(N features) tile.
// Results scattered: q[h][tok][d] (pre-scaled 1/8), k[h][tok][d], v_t[h][d][tok].
__global__ void __launch_bounds__(32)
qkv_gemm(const unsigned short* __restrict__ xb,
         const unsigned short* __restrict__ wb,
         unsigned short* __restrict__ qh,
         unsigned short* __restrict__ kh,
         unsigned short* __restrict__ vt) {
    const int lane = threadIdx.x & 31;
    const int col  = lane & 15;   // A row / B,C,D column index within 16-tile
    const int h16  = lane >> 4;   // half-wave selector
    const int m0   = blockIdx.x * 16;
    const int n0   = blockIdx.y * 64;

    v8f acc[4] = {};
    const unsigned short* arow = xb + (size_t)(m0 + col) * IN_DIM;
    for (int kk = 0; kk < IN_DIM; kk += 32) {
        Frag a;  // A 16x32 bf16: elems 0..15 -> K = h16*16.. (contiguous)
        a.q[0] = *(const uint4*)(arow + kk + h16 * 16);
        a.q[1] = *(const uint4*)(arow + kk + h16 * 16 + 8);
#pragma unroll
        for (int nb = 0; nb < 4; ++nb) {
            // B 32x16: col = lane%16, 16 contiguous K per half-wave
            const unsigned short* brow =
                wb + (size_t)(n0 + nb * 16 + col) * IN_DIM + kk + h16 * 16;
            Frag b;
            b.q[0] = *(const uint4*)(brow);
            b.q[1] = *(const uint4*)(brow + 8);
            acc[nb] = wmma_bf16(a, b, acc[nb]);
        }
    }
#pragma unroll
    for (int nb = 0; nb < 4; ++nb) {
        const int e    = n0 + nb * 16 + col;
        const int sel  = e >> 10;
        const int eh   = e & 1023;
        const int head = eh >> 6;
        const int d    = eh & 63;
#pragma unroll
        for (int v = 0; v < 8; ++v) {
            const int tok = m0 + v + 8 * h16;  // D layout: M = vgpr + 8*(lane/16)
            const float val = acc[nb][v];
            if (sel == 0)
                qh[((size_t)head * TOKENS + tok) * HDIM + d] = f2bf(val * 0.125f);
            else if (sel == 1)
                kh[((size_t)head * TOKENS + tok) * HDIM + d] = f2bf(val);
            else
                vt[((size_t)head * HDIM + d) * TOKENS + tok] = f2bf(val);
        }
    }
}

// ---------------- streaming-softmax attention ----------------
// One wave per (16-query tile, head). Online softmax over 4096 keys, 32 at a time.
__global__ void __launch_bounds__(32)
attn_kernel(const unsigned short* __restrict__ qh,
            const unsigned short* __restrict__ kh,
            const unsigned short* __restrict__ vt,
            unsigned short* __restrict__ aout) {
    __shared__ unsigned short lds_p[16 * 32];  // P tile, [row][key] bf16

    const int lane = threadIdx.x & 31;
    const int col  = lane & 15;
    const int h16  = lane >> 4;
    const int q0   = blockIdx.x * 16;
    const int head = blockIdx.y;

    // Q A-fragments, K-dim = head dim split 0..31 / 32..63
    const unsigned short* qrow = qh + ((size_t)head * TOKENS + q0 + col) * HDIM;
    Frag a0, a1;
    a0.q[0] = *(const uint4*)(qrow + h16 * 16);
    a0.q[1] = *(const uint4*)(qrow + 8 + h16 * 16);
    a1.q[0] = *(const uint4*)(qrow + 32 + h16 * 16);
    a1.q[1] = *(const uint4*)(qrow + 40 + h16 * 16);

    float m[8], l[8];
#pragma unroll
    for (int v = 0; v < 8; ++v) { m[v] = -1e30f; l[v] = 0.f; }
    v8f O[4] = {};

    for (int j0 = 0; j0 < TOKENS; j0 += 32) {
        // ---- scores: S = Q K^T, two 16-key tiles (B col = key, contiguous K = d)
        v8f s0 = {}, s1 = {};
        {
            const unsigned short* kr0 =
                kh + ((size_t)head * TOKENS + j0 + col) * HDIM + h16 * 16;
            Frag b;
            b.q[0] = *(const uint4*)(kr0);       b.q[1] = *(const uint4*)(kr0 + 8);
            s0 = wmma_bf16(a0, b, s0);
            b.q[0] = *(const uint4*)(kr0 + 32);  b.q[1] = *(const uint4*)(kr0 + 40);
            s0 = wmma_bf16(a1, b, s0);
        }
        {
            const unsigned short* kr1 =
                kh + ((size_t)head * TOKENS + j0 + 16 + col) * HDIM + h16 * 16;
            Frag b;
            b.q[0] = *(const uint4*)(kr1);       b.q[1] = *(const uint4*)(kr1 + 8);
            s1 = wmma_bf16(a0, b, s1);
            b.q[0] = *(const uint4*)(kr1 + 32);  b.q[1] = *(const uint4*)(kr1 + 40);
            s1 = wmma_bf16(a1, b, s1);
        }

        // ---- row max across the 16 lanes holding this row's columns
        float tm[8];
#pragma unroll
        for (int v = 0; v < 8; ++v) tm[v] = fmaxf(s0[v], s1[v]);
#pragma unroll
        for (int xr = 1; xr < 16; xr <<= 1)
#pragma unroll
            for (int v = 0; v < 8; ++v)
                tm[v] = fmaxf(tm[v], __shfl_xor(tm[v], xr, 16));

        float p0[8], p1[8], rs[8];
#pragma unroll
        for (int v = 0; v < 8; ++v) {
            const float nm   = fmaxf(m[v], tm[v]);
            const float corr = __expf(m[v] - nm);
            m[v] = nm;
            l[v] *= corr;
            p0[v] = __expf(s0[v] - nm);
            p1[v] = __expf(s1[v] - nm);
            rs[v] = p0[v] + p1[v];
            O[0][v] *= corr; O[1][v] *= corr; O[2][v] *= corr; O[3][v] *= corr;
        }
#pragma unroll
        for (int xr = 1; xr < 16; xr <<= 1)
#pragma unroll
            for (int v = 0; v < 8; ++v)
                rs[v] += __shfl_xor(rs[v], xr, 16);
#pragma unroll
        for (int v = 0; v < 8; ++v) l[v] += rs[v];

        // ---- D-layout -> A-layout transpose of P via LDS
#pragma unroll
        for (int v = 0; v < 8; ++v) {
            const int row = v + 8 * h16;
            lds_p[row * 32 + col]      = f2bf(p0[v]);
            lds_p[row * 32 + 16 + col] = f2bf(p1[v]);
        }
        __syncthreads();
        Frag pa;
        const unsigned short* pr = lds_p + col * 32;
        pa.q[0] = *(const uint4*)(pr + h16 * 16);
        pa.q[1] = *(const uint4*)(pr + 8 + h16 * 16);

        // ---- O += P V  (B col = d, contiguous K = keys via transposed v)
#pragma unroll
        for (int ch = 0; ch < 4; ++ch) {
            const unsigned short* vr =
                vt + ((size_t)head * HDIM + ch * 16 + col) * TOKENS + j0 + h16 * 16;
            Frag vb;
            vb.q[0] = *(const uint4*)(vr);
            vb.q[1] = *(const uint4*)(vr + 8);
            O[ch] = wmma_bf16(pa, vb, O[ch]);
        }
        __syncthreads();
    }

    // ---- finalize: O / l, scatter into [tok][head*64+d]
    float inv[8];
#pragma unroll
    for (int v = 0; v < 8; ++v) inv[v] = 1.f / l[v];
#pragma unroll
    for (int ch = 0; ch < 4; ++ch) {
#pragma unroll
        for (int v = 0; v < 8; ++v) {
            const int tok = q0 + v + 8 * h16;
            const int d   = ch * 16 + col;
            aout[(size_t)tok * EMB + head * HDIM + d] = f2bf(O[ch][v] * inv[v]);
        }
    }
}

// ---------------- output projection: y = aout @ out_w^T + out_b ----------------
__global__ void __launch_bounds__(32)
proj_gemm(const unsigned short* __restrict__ ab,
          const unsigned short* __restrict__ wb,
          const float* __restrict__ bias,
          float* __restrict__ out) {
    const int lane = threadIdx.x & 31;
    const int col  = lane & 15;
    const int h16  = lane >> 4;
    const int m0   = blockIdx.x * 16;
    const int n0   = blockIdx.y * 64;

    v8f acc[4] = {};
    const unsigned short* arow = ab + (size_t)(m0 + col) * EMB;
    for (int kk = 0; kk < EMB; kk += 32) {
        Frag a;
        a.q[0] = *(const uint4*)(arow + kk + h16 * 16);
        a.q[1] = *(const uint4*)(arow + kk + h16 * 16 + 8);
#pragma unroll
        for (int nb = 0; nb < 4; ++nb) {
            const unsigned short* brow =
                wb + (size_t)(n0 + nb * 16 + col) * EMB + kk + h16 * 16;
            Frag b;
            b.q[0] = *(const uint4*)(brow);
            b.q[1] = *(const uint4*)(brow + 8);
            acc[nb] = wmma_bf16(a, b, acc[nb]);
        }
    }
#pragma unroll
    for (int nb = 0; nb < 4; ++nb) {
        const int f = n0 + nb * 16 + col;
        const float bv = bias[f];
#pragma unroll
        for (int v = 0; v < 8; ++v) {
            const int tok = m0 + v + 8 * h16;
            out[(size_t)tok * EMB + f] = acc[nb][v] + bv;
        }
    }
}

extern "C" void kernel_launch(void* const* d_in, const int* in_sizes, int n_in,
                              void* d_out, int out_size, void* d_ws, size_t ws_size,
                              hipStream_t stream) {
    (void)in_sizes; (void)n_in; (void)out_size; (void)ws_size;
    const float* x     = (const float*)d_in[0];
    const float* qkv_w = (const float*)d_in[1];
    const float* out_w = (const float*)d_in[2];
    const float* out_b = (const float*)d_in[3];
    float* out = (float*)d_out;

    unsigned short* xb    = (unsigned short*)d_ws;            // 4096*1024
    unsigned short* wqkvb = xb + (size_t)TOKENS * IN_DIM;     // 3072*1024
    unsigned short* woutb = wqkvb + (size_t)QKV_DIM * IN_DIM; // 1024*1024
    unsigned short* qh    = woutb + (size_t)EMB * EMB;        // 16*4096*64
    unsigned short* kh    = qh + (size_t)NHEAD * TOKENS * HDIM;
    unsigned short* vt    = kh + (size_t)NHEAD * TOKENS * HDIM;
    unsigned short* aout  = vt + (size_t)NHEAD * TOKENS * HDIM; // 4096*1024

    const int nx = TOKENS * IN_DIM;
    const int nq = QKV_DIM * IN_DIM;
    const int no = EMB * EMB;
    cvt_kernel<<<(nx + 255) / 256, 256, 0, stream>>>(x, xb, nx);
    cvt_kernel<<<(nq + 255) / 256, 256, 0, stream>>>(qkv_w, wqkvb, nq);
    cvt_kernel<<<(no + 255) / 256, 256, 0, stream>>>(out_w, woutb, no);

    qkv_gemm<<<dim3(TOKENS / 16, QKV_DIM / 64), 32, 0, stream>>>(xb, wqkvb, qh, kh, vt);
    attn_kernel<<<dim3(TOKENS / 16, NHEAD), 32, 0, stream>>>(qh, kh, vt, aout);
    proj_gemm<<<dim3(TOKENS / 16, EMB / 64), 32, 0, stream>>>(aout, woutb, out_b, out);
}
